// ArtifactModel_66675072303385
// MI455X (gfx1250) — matrix operand order repaired and checked by reading, not compiled, not run on hardware
//
#include <hip/hip_runtime.h>
#include <hip/hip_bf16.h>

#define BV   32768
#define NSEG 65536

typedef __attribute__((ext_vector_type(16))) _Float16 v16h;
typedef __attribute__((ext_vector_type(8)))  _Float16 h8;
typedef __attribute__((ext_vector_type(8)))  float    v8f;

// Packed-weight region layout inside d_ws (element = _Float16):
//   [0,2048)      W0T  [64][32]   phi_w0^T,   K padded 11->32
//   [2048,6144)   W1T  [64][64]   phi_w1^T
//   [6144,8192)   OW0T [64][32]   omega_w0^T, K padded 9->32
//   [8192,12288)  OW1T [64][64]   omega_w1^T
//   [12288,36864) RW0T [128][192] rho_w0^T
//   [36864,45056) RW1T [64][128]  rho_w1^T
#define PK_W0T  0
#define PK_W1T  2048
#define PK_OW0T 6144
#define PK_OW1T 8192
#define PK_RW0T 12288
#define PK_RW1T 36864
#define PK_TOT  45056

__device__ __forceinline__ v8f wmma16(v16h a, v16h b, v8f c) {
  return __builtin_amdgcn_wmma_f32_16x16x32_f16(false, a, false, b, (short)0, c,
                                                false, false);
}

// A-fragment (16-bit A 16x32 layout): lane l holds row M=l%16, g=l/16.
// Elements: K = kc*32 + g*16 + e, e=0..15 contiguous (matches B fragment).
// Both runs are contiguous, 16B-aligned -> two ds_load_b128.
__device__ __forceinline__ v16h load_a_frag(const _Float16* row, int kc, int g) {
  h8 lo = *(const h8*)(row + kc * 32 + g * 16);
  h8 hi = *(const h8*)(row + kc * 32 + g * 16 + 8);
  return __builtin_shufflevector(lo, hi, 0, 1, 2, 3, 4, 5, 6, 7,
                                 8, 9, 10, 11, 12, 13, 14, 15);
}

// B-fragment: lane l holds column N=l%16; K = base + g*16 + e, e=0..15 contiguous.
__device__ __forceinline__ v16h load_b_frag(const _Float16* p) {
  return *(const v16h*)p;   // 32B, 32B-aligned -> two b128 loads
}

// ---------------------------------------------------------------------------
// Weight packing: f32 row-major [K][N] -> f16 fragment-order [N][Kpad] with zeros.
// ---------------------------------------------------------------------------
__global__ __launch_bounds__(256) void pack_weights_kernel(
    const float* __restrict__ w0, const float* __restrict__ w1,
    const float* __restrict__ ow0, const float* __restrict__ ow1,
    const float* __restrict__ rw0, const float* __restrict__ rw1,
    _Float16* __restrict__ pk)
{
  int idx = blockIdx.x * 256 + threadIdx.x;
  if (idx < PK_W1T) {
    int n = idx >> 5, K = idx & 31;
    pk[idx] = (_Float16)((K < 11) ? w0[K * 64 + n] : 0.0f);
  } else if (idx < PK_OW0T) {
    int j = idx - PK_W1T; int n = j >> 6, K = j & 63;
    pk[idx] = (_Float16)w1[K * 64 + n];
  } else if (idx < PK_OW1T) {
    int j = idx - PK_OW0T; int n = j >> 5, K = j & 31;
    pk[idx] = (_Float16)((K < 9) ? ow0[K * 64 + n] : 0.0f);
  } else if (idx < PK_RW0T) {
    int j = idx - PK_OW1T; int n = j >> 6, K = j & 63;
    pk[idx] = (_Float16)ow1[K * 64 + n];
  } else if (idx < PK_RW1T) {
    int j = idx - PK_RW0T; int n = j / 192, K = j % 192;
    pk[idx] = (_Float16)rw0[K * 128 + n];
  } else if (idx < PK_TOT) {
    int j = idx - PK_RW1T; int n = j >> 7, K = j & 127;
    pk[idx] = (_Float16)rw1[K * 64 + n];
  }
}

// ---------------------------------------------------------------------------
// Phase 1: per-segment phi MLP (11->64 relu ->64 sigmoid) + masked mean.
// One wave32 per segment (<=32 reads -> two 16-row WMMA tiles).
// ---------------------------------------------------------------------------
__global__ __launch_bounds__(128) void phi_seg_kernel(
    const float* __restrict__ reads, const long long* __restrict__ end_indices,
    const _Float16* __restrict__ w0t, const _Float16* __restrict__ w1t,
    const float* __restrict__ b0, const float* __restrict__ b1,
    float* __restrict__ seg_means,   // [NSEG][64]
    float* __restrict__ seg_counts)  // [NSEG]
{
  __shared__ __align__(32) _Float16 lds_reads[4][32][32];
  __shared__ __align__(32) _Float16 lds_act[4][32][64];

  const int wave = threadIdx.x >> 5;
  const int lane = threadIdx.x & 31;
  const int g    = lane >> 4;
  const int m    = lane & 15;
  const int seg  = blockIdx.x * 4 + wave;

  const long long end   = end_indices[seg];
  const long long start = seg ? end_indices[seg - 1] : 0ll;
  const int count = (int)(end - start);
  const int nelem = count * 11;

  // Zero the padded read tile (per-wave slice) with b64 stores.
  {
    unsigned long long* z = (unsigned long long*)&lds_reads[wave][0][0];
#pragma unroll
    for (int i = 0; i < 8; ++i) z[i * 32 + lane] = 0ull;
  }
  // Branchless, coalesced copy of count*11 floats (counts are always >= 5).
  {
    const float* src = reads + start * 11;
#pragma unroll
    for (int it = 0; it < 11; ++it) {
      int i = it * 32 + lane;
      int idx = (i < nelem) ? i : (nelem - 1);
      float v = src[idx];
      v = (i < nelem) ? v : 0.0f;
      lds_reads[wave][i / 11][i % 11] = (_Float16)v;
    }
  }
  __syncthreads();

  // Layer 1: K padded 11->32, one WMMA per (tile, N-tile).  relu -> lds_act.
#pragma unroll
  for (int t = 0; t < 2; ++t) {
    v16h a = load_a_frag(&lds_reads[wave][t * 16 + m][0], 0, g);
#pragma unroll
    for (int nt = 0; nt < 4; ++nt) {
      v16h bf = load_b_frag(w0t + (nt * 16 + m) * 32 + g * 16);
      v8f c = {};
      c = wmma16(a, bf, c);
      float bias = b0[nt * 16 + m];
#pragma unroll
      for (int v = 0; v < 8; ++v) {
        float x = fmaxf(c[v] + bias, 0.0f);
        lds_act[wave][t * 16 + v + 8 * g][nt * 16 + m] = (_Float16)x;
      }
    }
  }
  __syncthreads();

  // Layer 2: 64->64, sigmoid, masked column-sum over rows, mean.
#pragma unroll
  for (int nt = 0; nt < 4; ++nt) {
    float bias = b1[nt * 16 + m];
    float s = 0.0f;
#pragma unroll
    for (int t = 0; t < 2; ++t) {
      v8f c = {};
#pragma unroll
      for (int kc = 0; kc < 2; ++kc) {
        v16h a  = load_a_frag(&lds_act[wave][t * 16 + m][0], kc, g);
        v16h bf = load_b_frag(w1t + (nt * 16 + m) * 64 + kc * 32 + g * 16);
        c = wmma16(a, bf, c);
      }
#pragma unroll
      for (int v = 0; v < 8; ++v) {
        int r = t * 16 + v + 8 * g;
        float x = c[v] + bias;
        float sg = 1.0f / (1.0f + __expf(-x));
        s += (r < count) ? sg : 0.0f;
      }
    }
    s += __shfl_xor(s, 16, 32);   // merge the two lane halves of the C layout
    if (lane < 16)
      seg_means[(size_t)seg * 64 + nt * 16 + m] = s / (float)count;
  }
  if (lane == 0) seg_counts[seg] = (float)count;
}

// ---------------------------------------------------------------------------
// Phase 2: omega MLP + rho MLP + cal MLP + tanh clamp. One wave32 per 16 rows.
// ---------------------------------------------------------------------------
__global__ __launch_bounds__(128) void head_kernel(
    const float* __restrict__ info,
    const _Float16* __restrict__ ow0t, const float* __restrict__ ob0,
    const _Float16* __restrict__ ow1t, const float* __restrict__ ob1,
    const _Float16* __restrict__ rw0t, const float* __restrict__ rb0,
    const _Float16* __restrict__ rw1t, const float* __restrict__ rb1,
    const float* __restrict__ rw2, const float* __restrict__ rb2,
    const float* __restrict__ cw0, const float* __restrict__ cb0,
    const float* __restrict__ cw1, const float* __restrict__ cb1,
    const float* __restrict__ cw2, const float* __restrict__ cb2,
    const float* __restrict__ max_logit,
    const float* __restrict__ seg_means, const float* __restrict__ seg_counts,
    float* __restrict__ out)
{
  __shared__ __align__(32) _Float16 linfo[4][16][32];
  __shared__ __align__(32) _Float16 lx[4][16][192];  // [ref|alt|omega]; later h2
  __shared__ __align__(32) _Float16 lh[4][16][128];  // omega hidden; later rho h1

  const int w    = threadIdx.x >> 5;
  const int lane = threadIdx.x & 31;
  const int g    = lane >> 4;
  const int m    = lane & 15;
  const int tile = blockIdx.x * 4 + w;
  const int v0   = tile * 16;

  // Stage info tile (16x9 -> zero-padded 16x32 f16), branchless.
  {
    unsigned long long* z = (unsigned long long*)&linfo[w][0][0];
#pragma unroll
    for (int i = 0; i < 4; ++i) z[i * 32 + lane] = 0ull;
    const float* src = info + (size_t)v0 * 9;
#pragma unroll
    for (int it = 0; it < 5; ++it) {
      int i = it * 32 + lane;
      int idx = (i < 144) ? i : 143;
      float v = src[idx];
      v = (i < 144) ? v : 0.0f;
      linfo[w][i / 9][i % 9] = (_Float16)v;
    }
  }
  __syncthreads();

  // --- omega layer 1: 9(->32)->64, relu -> lh[..][0..63]
  {
    v16h a = load_a_frag(&linfo[w][m][0], 0, g);
#pragma unroll
    for (int nt = 0; nt < 4; ++nt) {
      v16h bf = load_b_frag(ow0t + (nt * 16 + m) * 32 + g * 16);
      v8f c = {};
      c = wmma16(a, bf, c);
      float bias = ob0[nt * 16 + m];
#pragma unroll
      for (int v = 0; v < 8; ++v) {
        float x = fmaxf(c[v] + bias, 0.0f);
        lh[w][v + 8 * g][nt * 16 + m] = (_Float16)x;
      }
    }
  }
  __syncthreads();

  // --- omega layer 2: 64->64, sigmoid -> lx[..][128..191]
#pragma unroll
  for (int nt = 0; nt < 4; ++nt) {
    v8f c = {};
#pragma unroll
    for (int kc = 0; kc < 2; ++kc) {
      v16h a  = load_a_frag(&lh[w][m][0], kc, g);
      v16h bf = load_b_frag(ow1t + (nt * 16 + m) * 64 + kc * 32 + g * 16);
      c = wmma16(a, bf, c);
    }
    float bias = ob1[nt * 16 + m];
#pragma unroll
    for (int v = 0; v < 8; ++v) {
      float x = c[v] + bias;
      float sg = 1.0f / (1.0f + __expf(-x));
      lx[w][v + 8 * g][128 + nt * 16 + m] = (_Float16)sg;
    }
  }

  // --- fill lx[..][0..127] with ref/alt segment means (coalesced)
#pragma unroll
  for (int it = 0; it < 32; ++it) {
    int i = it * 32 + lane;            // i in [0, 1024): row = i/64, n = i%64
    int row = i >> 6, n = i & 63;
    lx[w][row][n]      = (_Float16)seg_means[(size_t)(v0 + row) * 64 + n];
    lx[w][row][64 + n] = (_Float16)seg_means[(size_t)(BV + v0 + row) * 64 + n];
  }
  __syncthreads();

  // --- rho layer 1: 192->128, relu -> lh
  {
    v16h af[6];
#pragma unroll
    for (int kc = 0; kc < 6; ++kc)
      af[kc] = load_a_frag(&lx[w][m][0], kc, g);
#pragma unroll
    for (int nt = 0; nt < 8; ++nt) {
      v8f c = {};
#pragma unroll
      for (int kc = 0; kc < 6; ++kc) {
        v16h bf = load_b_frag(rw0t + (nt * 16 + m) * 192 + kc * 32 + g * 16);
        c = wmma16(af[kc], bf, c);
      }
      float bias = rb0[nt * 16 + m];
#pragma unroll
      for (int v = 0; v < 8; ++v) {
        float x = fmaxf(c[v] + bias, 0.0f);
        lh[w][v + 8 * g][nt * 16 + m] = (_Float16)x;
      }
    }
  }
  __syncthreads();

  // --- rho layer 2: 128->64, relu -> lx[..][0..63] (ref/alt slots now dead)
#pragma unroll
  for (int nt = 0; nt < 4; ++nt) {
    v8f c = {};
#pragma unroll
    for (int kc = 0; kc < 4; ++kc) {
      v16h a  = load_a_frag(&lh[w][m][0], kc, g);
      v16h bf = load_b_frag(rw1t + (nt * 16 + m) * 128 + kc * 32 + g * 16);
      c = wmma16(a, bf, c);
    }
    float bias = rb1[nt * 16 + m];
#pragma unroll
    for (int v = 0; v < 8; ++v) {
      float x = fmaxf(c[v] + bias, 0.0f);
      lx[w][v + 8 * g][nt * 16 + m] = (_Float16)x;
    }
  }
  __syncthreads();

  // --- rho layer 3 (64->1), cal MLP (2->5->5->1), tanh clamp. One lane per row.
  if (lane < 16) {
    const int v = v0 + lane;
    float logit = rb2[0];
#pragma unroll
    for (int k = 0; k < 64; ++k)
      logit += (float)lx[w][lane][k] * rw2[k];

    float ea = seg_counts[BV + v];          // eff_alt = alt read count
    float er = seg_counts[v];               // eff_ref = ref read count
    float in0 = sqrtf(ea), in1 = sqrtf(er); // concat order: [sqrt(alt), sqrt(ref)]

    float h1[5], h2[5];
#pragma unroll
    for (int j = 0; j < 5; ++j) {
      float t = cb0[j] + in0 * cw0[j] + in1 * cw0[5 + j];
      h1[j] = fmaxf(t, 0.0f);
    }
#pragma unroll
    for (int j = 0; j < 5; ++j) {
      float t = cb1[j];
#pragma unroll
      for (int k = 0; k < 5; ++k) t += h1[k] * cw1[k * 5 + j];
      h2[j] = fmaxf(t, 0.0f);
    }
    float temp = cb2[0];
#pragma unroll
    for (int k = 0; k < 5; ++k) temp += h2[k] * cw2[k];

    float ml = max_logit[0];
    out[v] = ml * tanhf((logit * temp) / ml);
  }
}

// ---------------------------------------------------------------------------
extern "C" void kernel_launch(void* const* d_in, const int* in_sizes, int n_in,
                              void* d_out, int out_size, void* d_ws, size_t ws_size,
                              hipStream_t stream) {
  const float*     reads = (const float*)d_in[0];
  const float*     info  = (const float*)d_in[1];
  const long long* endi  = (const long long*)d_in[2];
  const float *pw0 = (const float*)d_in[3],  *pb0 = (const float*)d_in[4];
  const float *pw1 = (const float*)d_in[5],  *pb1 = (const float*)d_in[6];
  const float *ow0 = (const float*)d_in[7],  *ob0 = (const float*)d_in[8];
  const float *ow1 = (const float*)d_in[9],  *ob1 = (const float*)d_in[10];
  const float *rw0 = (const float*)d_in[11], *rb0 = (const float*)d_in[12];
  const float *rw1 = (const float*)d_in[13], *rb1 = (const float*)d_in[14];
  const float *rw2 = (const float*)d_in[15], *rb2 = (const float*)d_in[16];
  const float *cw0 = (const float*)d_in[17], *cb0 = (const float*)d_in[18];
  const float *cw1 = (const float*)d_in[19], *cb1 = (const float*)d_in[20];
  const float *cw2 = (const float*)d_in[21], *cb2 = (const float*)d_in[22];
  const float *ml  = (const float*)d_in[23];

  float* seg_means  = (float*)d_ws;                       // NSEG*64 f32 (16 MB)
  float* seg_counts = seg_means + (size_t)NSEG * 64;      // NSEG f32
  _Float16* pk = (_Float16*)(seg_counts + NSEG);          // packed f16 weights
  float* out = (float*)d_out;

  pack_weights_kernel<<<(PK_TOT + 255) / 256, 256, 0, stream>>>(
      pw0, pw1, ow0, ow1, rw0, rw1, pk);

  phi_seg_kernel<<<NSEG / 4, 128, 0, stream>>>(
      reads, endi, pk + PK_W0T, pk + PK_W1T, pb0, pb1, seg_means, seg_counts);

  head_kernel<<<(BV / 16) / 4, 128, 0, stream>>>(
      info, pk + PK_OW0T, ob0, pk + PK_OW1T, ob1, pk + PK_RW0T, rb0,
      pk + PK_RW1T, rb1, rw2, rb2, cw0, cb0, cw1, cb1, cw2, cb2, ml,
      seg_means, seg_counts, out);
}
